// CrossAttention_80599356277123
// MI455X (gfx1250) — matrix operand order repaired and checked by reading, not compiled, not run on hardware
//
#include <hip/hip_runtime.h>
#include <hip/hip_bf16.h>

#define BATCH 2
#define SEQ   2048
#define DIMC  1024
#define NHEADS 16
#define HD    64

typedef __attribute__((ext_vector_type(16))) __bf16 v16bf;
typedef __attribute__((ext_vector_type(8)))  float  v8f;

union FragBF {
    v16bf v;
    unsigned u[8];
    uint4 q[2];
    __bf16 h[16];
};

__device__ inline v8f wmma_bf16(const FragBF& a, const FragBF& b, v8f c) {
    // D = A(16x32 bf16) x B(32x16 bf16) + C(16x16 f32)
    return __builtin_amdgcn_wmma_f32_16x16x32_bf16(
        false, a.v, false, b.v, (short)0, c, false, false);
}

// ---------------------------------------------------------------------------
// Vectorized f32 -> bf16 (4 elements / thread)
// ---------------------------------------------------------------------------
__global__ void ca_cvt_bf16(const float4* __restrict__ in, uint2* __restrict__ out, int n4) {
    int i = blockIdx.x * blockDim.x + threadIdx.x;
    if (i < n4) {
        float4 f = in[i];
        union { __bf16 h[4]; uint2 u; } p;
        p.h[0] = (__bf16)f.x; p.h[1] = (__bf16)f.y;
        p.h[2] = (__bf16)f.z; p.h[3] = (__bf16)f.w;
        out[i] = p.u;
    }
}

// LDS-tiled transposing convert: in[K][N] f32 -> out[N][K] bf16 (coalesced both ways)
__global__ void ca_tcvt_bf16(const float* __restrict__ in, __bf16* __restrict__ out,
                             int N, int K) {
    __shared__ float t[32][33];
    int n0 = blockIdx.x * 32, k0 = blockIdx.y * 32;
#pragma unroll
    for (int i = 0; i < 4; ++i)
        t[threadIdx.y + 8 * i][threadIdx.x] =
            in[(size_t)(k0 + threadIdx.y + 8 * i) * N + n0 + threadIdx.x];
    __syncthreads();
#pragma unroll
    for (int i = 0; i < 4; ++i)
        out[(size_t)(n0 + threadIdx.y + 8 * i) * K + k0 + threadIdx.x] =
            (__bf16)t[threadIdx.x][threadIdx.y + 8 * i];
}

// ---------------------------------------------------------------------------
// 32x32 tile GEMM core: one wave -> 2x2 grid of 16x16 WMMA tiles.
// A[M,K] bf16 row-major, BT[N,K] bf16 row-major (pre-transposed weights).
// 4 fragment loads feed 4 WMMAs per 32-deep k-chunk (2 B128 loads / WMMA).
// ---------------------------------------------------------------------------
struct Acc4 { v8f a00, a01, a10, a11; };

template <int KDIM>
__device__ inline Acc4 gemm_core32(const __bf16* __restrict__ A0, const __bf16* __restrict__ A1,
                                   const __bf16* __restrict__ B0, const __bf16* __restrict__ B1,
                                   int hi8, int hi16) {
    Acc4 acc;
    acc.a00 = {}; acc.a01 = {}; acc.a10 = {}; acc.a11 = {};
#pragma unroll 2
    for (int k0 = 0; k0 < KDIM; k0 += 32) {
        FragBF a0, a1, b0, b1;
        a0.q[0] = *(const uint4*)(A0 + k0 + hi8);
        a0.q[1] = *(const uint4*)(A0 + k0 + 16 + hi8);
        a1.q[0] = *(const uint4*)(A1 + k0 + hi8);
        a1.q[1] = *(const uint4*)(A1 + k0 + 16 + hi8);
        b0.q[0] = *(const uint4*)(B0 + k0 + hi8);
        b0.q[1] = *(const uint4*)(B0 + k0 + 16 + hi8);
        b1.q[0] = *(const uint4*)(B1 + k0 + hi8);
        b1.q[1] = *(const uint4*)(B1 + k0 + 16 + hi8);
        acc.a00 = wmma_bf16(a0, b0, acc.a00);
        acc.a01 = wmma_bf16(a0, b1, acc.a01);
        acc.a10 = wmma_bf16(a1, b0, acc.a10);
        acc.a11 = wmma_bf16(a1, b1, acc.a11);
    }
    return acc;
}

// Q projection: q = (x @ Wq + bq) * hd^-0.5, stored bf16 as [B][H][N][hd]
__global__ void ca_gemm_q(const __bf16* __restrict__ A, const __bf16* __restrict__ BT,
                          const float* __restrict__ bias, __bf16* __restrict__ qout) {
    int wave = blockIdx.x * (blockDim.x >> 5) + (threadIdx.x >> 5);
    int lane = threadIdx.x & 31;
    int nm = lane & 15, hi = lane >> 4, hi8 = hi << 3, hi16 = hi << 4;
    int row0 = (wave >> 5) << 5;     // tilesN = DIM/32 = 32
    int col0 = (wave & 31) << 5;
    Acc4 acc = gemm_core32<DIMC>(A + (size_t)(row0 + nm) * DIMC,
                                 A + (size_t)(row0 + 16 + nm) * DIMC,
                                 BT + (size_t)(col0 + nm) * DIMC,
                                 BT + (size_t)(col0 + 16 + nm) * DIMC, hi8, hi16);
    const v8f* accs[4] = {&acc.a00, &acc.a01, &acc.a10, &acc.a11};
#pragma unroll
    for (int ti = 0; ti < 2; ++ti)
#pragma unroll
        for (int tj = 0; tj < 2; ++tj) {
            const v8f& a = *accs[ti * 2 + tj];
            int col = col0 + tj * 16 + nm;
            float bv = bias[col];
            int h = col >> 6, d = col & 63;
#pragma unroll
            for (int r = 0; r < 8; ++r) {
                int rr = row0 + ti * 16 + hi8 + r;        // global row in [0, B*N)
                int b = rr >> 11, nn = rr & 2047;
                size_t o = (((size_t)(b * NHEADS + h)) * SEQ + nn) * HD + d;
                qout[o] = (__bf16)((a[r] + bv) * 0.125f);
            }
        }
}

// KV projection: kv = y @ Wkv + bkv.  K -> [B][H][N2][hd], V -> [B][H][hd][N2]
__global__ void ca_gemm_kv(const __bf16* __restrict__ A, const __bf16* __restrict__ BT,
                           const float* __restrict__ bias,
                           __bf16* __restrict__ kout, __bf16* __restrict__ vout) {
    int wave = blockIdx.x * (blockDim.x >> 5) + (threadIdx.x >> 5);
    int lane = threadIdx.x & 31;
    int nm = lane & 15, hi = lane >> 4, hi8 = hi << 3, hi16 = hi << 4;
    int row0 = (wave >> 6) << 5;     // tilesN = 2*DIM/32 = 64
    int col0 = (wave & 63) << 5;
    Acc4 acc = gemm_core32<DIMC>(A + (size_t)(row0 + nm) * DIMC,
                                 A + (size_t)(row0 + 16 + nm) * DIMC,
                                 BT + (size_t)(col0 + nm) * DIMC,
                                 BT + (size_t)(col0 + 16 + nm) * DIMC, hi8, hi16);
    const v8f* accs[4] = {&acc.a00, &acc.a01, &acc.a10, &acc.a11};
#pragma unroll
    for (int ti = 0; ti < 2; ++ti)
#pragma unroll
        for (int tj = 0; tj < 2; ++tj) {
            const v8f& a = *accs[ti * 2 + tj];
            int col = col0 + tj * 16 + nm;
            float bv = bias[col];
            int s = col >> 10;                  // 0 = K, 1 = V (uniform per tile)
            int c2 = col & 1023;
            int h = c2 >> 6, d = c2 & 63;
#pragma unroll
            for (int r = 0; r < 8; ++r) {
                int rr = row0 + ti * 16 + hi8 + r;
                int b = rr >> 11, nn = rr & 2047;
                float val = a[r] + bv;
                if (s == 0)
                    kout[(((size_t)(b * NHEADS + h)) * SEQ + nn) * HD + d] = (__bf16)val;
                else
                    vout[(((size_t)(b * NHEADS + h)) * HD + d) * SEQ + nn] = (__bf16)val;
            }
        }
}

// O projection: out = ctx @ Wo + bo  (f32 output, row-major [B*N][DIM])
__global__ void ca_gemm_o(const __bf16* __restrict__ A, const __bf16* __restrict__ BT,
                          const float* __restrict__ bias, float* __restrict__ out) {
    int wave = blockIdx.x * (blockDim.x >> 5) + (threadIdx.x >> 5);
    int lane = threadIdx.x & 31;
    int nm = lane & 15, hi = lane >> 4, hi8 = hi << 3, hi16 = hi << 4;
    int row0 = (wave >> 5) << 5;
    int col0 = (wave & 31) << 5;
    Acc4 acc = gemm_core32<DIMC>(A + (size_t)(row0 + nm) * DIMC,
                                 A + (size_t)(row0 + 16 + nm) * DIMC,
                                 BT + (size_t)(col0 + nm) * DIMC,
                                 BT + (size_t)(col0 + 16 + nm) * DIMC, hi8, hi16);
    const v8f* accs[4] = {&acc.a00, &acc.a01, &acc.a10, &acc.a11};
#pragma unroll
    for (int ti = 0; ti < 2; ++ti)
#pragma unroll
        for (int tj = 0; tj < 2; ++tj) {
            const v8f& a = *accs[ti * 2 + tj];
            int col = col0 + tj * 16 + nm;
            float bv = bias[col];
#pragma unroll
            for (int r = 0; r < 8; ++r) {
                int rr = row0 + ti * 16 + hi8 + r;
                out[(size_t)rr * DIMC + col] = a[r] + bv;
            }
        }
}

// ---------------------------------------------------------------------------
// Flash attention, one wave per (b,h, 16-query tile).
// Computes S^T = K * Q^T so softmax rows (queries) map to lane columns;
// O^T = V^T * P^T accumulated online with f32 state.
// ---------------------------------------------------------------------------
__global__ void ca_attn(const __bf16* __restrict__ Q, const __bf16* __restrict__ K,
                        const __bf16* __restrict__ V, __bf16* __restrict__ ctx) {
    int wt = blockIdx.x * (blockDim.x >> 5) + (threadIdx.x >> 5);
    int lane = threadIdx.x & 31;
    int nq = lane & 15, hi = lane >> 4, hi8 = hi << 3, hi16 = hi << 4;
    int qt = wt & 127;             // N/16 = 128 query tiles
    int bh = wt >> 7;              // [0, B*H)

    // Q^T B-fragment for both 32-wide hd chunks (loaded once)
    const __bf16* Qrow = Q + ((size_t)bh * SEQ + qt * 16 + nq) * HD;
    FragBF qf0, qf1;
    qf0.q[0] = *(const uint4*)(Qrow + hi8);
    qf0.q[1] = *(const uint4*)(Qrow + 16 + hi8);
    qf1.q[0] = *(const uint4*)(Qrow + 32 + hi8);
    qf1.q[1] = *(const uint4*)(Qrow + 48 + hi8);

    const __bf16* Kb = K + (size_t)bh * SEQ * HD;
    const __bf16* Vb = V + (size_t)bh * HD * SEQ;

    float mstate = -3.0e38f, lstate = 0.0f;
    v8f o0 = {}, o1 = {}, o2 = {}, o3 = {};

    for (int j = 0; j < SEQ; j += 32) {
        const __bf16* K0 = Kb + (size_t)(j + nq) * HD;
        const __bf16* K1 = Kb + (size_t)(j + 16 + nq) * HD;
        FragBF a;
        v8f s0 = {}, s1 = {};
        a.q[0] = *(const uint4*)(K0 + hi8);      a.q[1] = *(const uint4*)(K0 + 16 + hi8);
        s0 = wmma_bf16(a, qf0, s0);
        a.q[0] = *(const uint4*)(K0 + 32 + hi8); a.q[1] = *(const uint4*)(K0 + 48 + hi8);
        s0 = wmma_bf16(a, qf1, s0);
        a.q[0] = *(const uint4*)(K1 + hi8);      a.q[1] = *(const uint4*)(K1 + 16 + hi8);
        s1 = wmma_bf16(a, qf0, s1);
        a.q[0] = *(const uint4*)(K1 + 32 + hi8); a.q[1] = *(const uint4*)(K1 + 48 + hi8);
        s1 = wmma_bf16(a, qf1, s1);

        // ---- online softmax over this lane's query column (Q pre-scaled) ----
        float tmax = s0[0];
#pragma unroll
        for (int r = 1; r < 8; ++r) tmax = fmaxf(tmax, s0[r]);
#pragma unroll
        for (int r = 0; r < 8; ++r) tmax = fmaxf(tmax, s1[r]);
        tmax = fmaxf(tmax, __shfl_xor(tmax, 16, 32));
        float mnew = fmaxf(mstate, tmax);
        float corr = __expf(mstate - mnew);
        float p0[8], p1[8], ssum = 0.0f;
#pragma unroll
        for (int r = 0; r < 8; ++r) {
            p0[r] = __expf(s0[r] - mnew);
            p1[r] = __expf(s1[r] - mnew);
            ssum += p0[r] + p1[r];
        }
        ssum += __shfl_xor(ssum, 16, 32);
        lstate = lstate * corr + ssum;
        mstate = mnew;

        // ---- build P^T B-fragment: lane needs keys hi*16 .. hi*16+15 ----
        FragBF pf;
#pragma unroll
        for (int t = 0; t < 8; ++t) {
            float q0 = __shfl_xor(p0[t], 16, 32);   // partner's tile0 values
            float q1 = __shfl_xor(p1[t], 16, 32);   // partner's tile1 values
            pf.h[t]     = (__bf16)(hi ? q1    : p0[t]);
            pf.h[8 + t] = (__bf16)(hi ? p1[t] : q0);
        }

        // ---- O^T += V^T(16d x 32k) * P^T(32k x 16q), 4 d-blocks ----
#pragma unroll
        for (int dblk = 0; dblk < 4; ++dblk) {
            const __bf16* Vrow = Vb + (size_t)(dblk * 16 + nq) * SEQ + j;
            FragBF vf;
            vf.q[0] = *(const uint4*)(Vrow + hi16);
            vf.q[1] = *(const uint4*)(Vrow + hi16 + 8);
            v8f* op = dblk == 0 ? &o0 : dblk == 1 ? &o1 : dblk == 2 ? &o2 : &o3;
            v8f t = *op;
#pragma unroll
            for (int r = 0; r < 8; ++r) t[r] *= corr;
            *op = wmma_bf16(vf, pf, t);
        }
    }

    float linv = 1.0f / lstate;
    int b = bh >> 4, h = bh & 15;
    size_t grow = ((size_t)b * SEQ + qt * 16 + nq) * DIMC + h * HD;
    v8f oarr[4] = {o0, o1, o2, o3};
#pragma unroll
    for (int dblk = 0; dblk < 4; ++dblk) {
#pragma unroll
        for (int rp = 0; rp < 4; ++rp) {
            union { __bf16 hh[2]; unsigned u; } pk;
            pk.hh[0] = (__bf16)(oarr[dblk][2 * rp] * linv);
            pk.hh[1] = (__bf16)(oarr[dblk][2 * rp + 1] * linv);
            *(unsigned*)(ctx + grow + dblk * 16 + hi8 + 2 * rp) = pk.u;
        }
    }
}

// ---------------------------------------------------------------------------
extern "C" void kernel_launch(void* const* d_in, const int* in_sizes, int n_in,
                              void* d_out, int out_size, void* d_ws, size_t ws_size,
                              hipStream_t stream) {
    const float* x   = (const float*)d_in[0];
    const float* y   = (const float*)d_in[1];
    const float* Wq  = (const float*)d_in[2];
    const float* bq  = (const float*)d_in[3];
    const float* Wkv = (const float*)d_in[4];
    const float* bkv = (const float*)d_in[5];
    const float* Wo  = (const float*)d_in[6];
    const float* bo  = (const float*)d_in[7];
    float* out = (float*)d_out;

    const int ACT = BATCH * SEQ * DIMC;            // 4194304
    __bf16* ws   = (__bf16*)d_ws;
    __bf16* xb   = ws;
    __bf16* yb   = xb   + ACT;
    __bf16* WqT  = yb   + ACT;
    __bf16* WkvT = WqT  + DIMC * DIMC;
    __bf16* WoT  = WkvT + DIMC * 2 * DIMC;
    __bf16* qs   = WoT  + DIMC * DIMC;
    __bf16* kb   = qs   + ACT;
    __bf16* vT   = kb   + ACT;
    __bf16* ctx  = vT   + ACT;

    // 1) precision conversion (+ coalesced tiled weight transpose)
    ca_cvt_bf16<<<ACT / 4 / 256, 256, 0, stream>>>((const float4*)x, (uint2*)xb, ACT / 4);
    ca_cvt_bf16<<<ACT / 4 / 256, 256, 0, stream>>>((const float4*)y, (uint2*)yb, ACT / 4);
    {
        dim3 blk(32, 8);
        ca_tcvt_bf16<<<dim3(DIMC / 32, DIMC / 32), blk, 0, stream>>>(Wq, WqT, DIMC, DIMC);
        ca_tcvt_bf16<<<dim3(2 * DIMC / 32, DIMC / 32), blk, 0, stream>>>(Wkv, WkvT, 2 * DIMC, DIMC);
        ca_tcvt_bf16<<<dim3(DIMC / 32, DIMC / 32), blk, 0, stream>>>(Wo, WoT, DIMC, DIMC);
    }

    // 2) projections (8 waves/block, one 32x32 tile per wave)
    ca_gemm_q <<<(128 * 32) / 8, 256, 0, stream>>>(xb, WqT, bq, qs);
    ca_gemm_kv<<<(128 * 64) / 8, 256, 0, stream>>>(yb, WkvT, bkv, kb, vT);

    // 3) flash attention: B*H*(N/16) = 4096 waves, 4 waves/block
    ca_attn<<<(BATCH * NHEADS * (SEQ / 16)) / 4, 128, 0, stream>>>(qs, kb, vT, ctx);

    // 4) output projection
    ca_gemm_o<<<(128 * 32) / 8, 256, 0, stream>>>(ctx, WoT, bo, out);
}
